// ClassicPINN_33157147525272
// MI455X (gfx1250) — matrix operand order repaired and checked by reading, not compiled, not run on hardware
//
#include <hip/hip_runtime.h>
#include <hip/hip_bf16.h>
#include <math.h>

typedef __attribute__((ext_vector_type(16))) _Float16 v16h;
typedef __attribute__((ext_vector_type(8)))  _Float16 v8h;
typedef __attribute__((ext_vector_type(8)))  float    v8f;
typedef __attribute__((ext_vector_type(4)))  float    v4f;

#define NPTS 1048576
// widths: 3 -> 8x7 -> 16x3 -> 32x3 -> 16 -> 3
// 18 WMMA blocks: layers 0..9 (1 blk), 10/11/12 (2 blks each), 13, 14
// Fragment record per block (2048 B): [A 16x32 f16, lane*32] [bias v8f, lane*32]

__device__ __forceinline__ float my_tanh(float x) {
#if __has_builtin(__builtin_amdgcn_tanhf)
  return __builtin_amdgcn_tanhf(x);
#else
  return tanhf(x);
#endif
}

// Issue 4 cache-scope loads for one fragment record; no wait here.
// Outputs are EARLY-CLOBBER so none can alias the address register pair
// (later loads in the clause still read %4 at issue time).
__device__ __forceinline__ void frag_issue(const char* p,
                                           v8h& alo, v8h& ahi, v4f& clo, v4f& chi) {
  asm volatile(
      "global_load_b128 %0, %4, off\n\t"
      "global_load_b128 %1, %4, off offset:16\n\t"
      "global_load_b128 %2, %4, off offset:1024\n\t"
      "global_load_b128 %3, %4, off offset:1040"
      : "=&v"(alo), "=&v"(ahi), "=&v"(clo), "=&v"(chi)
      : "v"(p));
}

// Single wait, ordered after the loads and before any use of the fragments.
// At every call site the only outstanding vector loads are these four, so
// waiting for loadcnt==0 is exact (no over-wait of the pipeline).
__device__ __forceinline__ void frag_wait(v8h& alo, v8h& ahi, v4f& clo, v4f& chi) {
  asm volatile("s_wait_loadcnt 0x0"
               : "+v"(alo), "+v"(ahi), "+v"(clo), "+v"(chi));
}

// ---------------- prep: swizzle weights/biases into WMMA operand layouts ----
__global__ __launch_bounds__(32) void pinn_prep(
    const float* W0, const float* W1, const float* W2, const float* W3,
    const float* W4, const float* W5, const float* W6, const float* W7,
    const float* W8, const float* W9, const float* W10, const float* W11,
    const float* W12, const float* W13, const float* W14,
    const float* b0, const float* b1, const float* b2, const float* b3,
    const float* b4, const float* b5, const float* b6, const float* b7,
    const float* b8, const float* b9, const float* b10, const float* b11,
    const float* b12, const float* b13, const float* b14,
    char* __restrict__ wsF)
{
  constexpr int WID[16] = {3,8,8,8,8,8,8,8,16,16,16,32,32,32,16,3};
  constexpr int LYR[18] = {0,1,2,3,4,5,6,7,8,9,10,10,11,11,12,12,13,14};
  constexpr int MBL[18] = {0,0,0,0,0,0,0,0,0,0,0, 1, 0, 1, 0, 1, 0, 0};
  const float* Ws[15] = {W0,W1,W2,W3,W4,W5,W6,W7,W8,W9,W10,W11,W12,W13,W14};
  const float* bs[15] = {b0,b1,b2,b3,b4,b5,b6,b7,b8,b9,b10,b11,b12,b13,b14};

  const int fb   = blockIdx.x;          // 0..17
  const int lane = threadIdx.x;         // 0..31
  const int L    = LYR[fb];
  const int mb   = MBL[fb];
  const int in   = WID[L];
  const int outw = WID[L + 1];
  const int m    = lane & 15;
  const int lh   = lane >> 4;
  const int M    = mb * 16 + m;
  const float* W = Ws[L];
  const float* b = bs[L];

  // A-fragment: 16x32 f16, lane=M, vgpr r halves: K = 16*lh + r*2 + h
  _Float16* ap = (_Float16*)(wsF + (size_t)fb * 2048 + (size_t)lane * 32);
  for (int r = 0; r < 8; ++r)
    for (int h = 0; h < 2; ++h) {
      int k = 16 * lh + (r << 1) + h;
      float v = (M < outw && k < in) ? W[M * in + k] : 0.0f;
      ap[r * 2 + h] = (_Float16)v;
    }

  // bias fragment in f32 C layout: vgpr r -> M = 16*mb + r + 8*lh
  float* bp = (float*)(wsF + (size_t)fb * 2048 + 1024 + (size_t)lane * 32);
  for (int r = 0; r < 8; ++r) {
    int MM = mb * 16 + r + 8 * lh;
    bp[r] = (MM < outw) ? b[MM] : 0.0f;
  }
}

// ---------------- main: 16 points per wave-tile, WMMA chain through 15 layers
// Flattened 18-block pipeline with 2-deep rotating fragment buffer: block
// fb+1's 4 loads are in flight during block fb's wmma/tanh/LDS round trip.
__global__ __launch_bounds__(256, 8) void pinn_main(
    const float* __restrict__ coords,
    const char* __restrict__ wsF,
    float* __restrict__ out)
{
  constexpr int BLYR[18] = {0,1,2,3,4,5,6,7,8,9,10,10,11,11,12,12,13,14};
  constexpr int BMBL[18] = {0,0,0,0,0,0,0,0,0,0,0, 1, 0, 1, 0, 1, 0, 0};
  __shared__ __align__(16) _Float16 hbuf[8][16][40];  // 8 waves x 16 pts x (32+8 pad)

  const int lane = threadIdx.x & 31;
  const int wave = threadIdx.x >> 5;
  const int n    = lane & 15;   // point within tile (for B and C/D layouts)
  const int lh   = lane >> 4;   // lane half
  _Float16* hrow = &hbuf[wave][n][0];

  const char* fragLane = wsF + (size_t)lane * 32;   // + fb*2048 per block

  const int waveGlobal = (blockIdx.x << 3) + wave;  // 1024 blocks * 8 waves = 8192

  // Fragment pipeline prologue: issue block 0 (addresses are tile-invariant).
  v8h pal[2], pah[2];
  v4f pcl[2], pch[2];
  frag_issue(fragLane, pal[0], pah[0], pcl[0], pch[0]);

  for (int t = 0; t < 8; ++t) {              // 8 tiles/wave -> 65536 tiles total
    const int tile = waveGlobal * 8 + t;
    const int p    = tile * 16 + n;

    // ---- tile init: h[n][0..31] = {x,y,z,0,...,0} as f16 -------------------
    v8h z;
#pragma unroll
    for (int i = 0; i < 8; ++i) z[i] = (_Float16)0.0f;
    v8h init0 = z;
    if (lh == 0) {
      const float* cp = coords + 3 * (size_t)p;
      init0[0] = (_Float16)cp[0];
      init0[1] = (_Float16)cp[1];
      init0[2] = (_Float16)cp[2];
    }
    *(v8h*)(hrow + 8 * lh)      = init0;  // feats 0-7 (coords) / 8-15 (zeros)
    *(v8h*)(hrow + 16 + 8 * lh) = z;      // feats 16-23 / 24-31 zeros

    // ---- 18 WMMA blocks (15 layers), fully unrolled ------------------------
    v16h bvec;
#pragma unroll
    for (int fb = 0; fb < 18; ++fb) {
      const int L   = BLYR[fb];
      const int mb  = BMBL[fb];
      const int cur = fb & 1;
      const int nxt = cur ^ 1;

      // Layer start: load B operand 32x16 f16 from LDS
      // (lane half lh holds K = 16*lh .. 16*lh+15 for column n)
      if (mb == 0)
        bvec = *(const v16h*)(hrow + 16 * lh);

      // Retire this block's fragment loads (exactly 4 outstanding here).
      frag_wait(pal[cur], pah[cur], pcl[cur], pch[cur]);
      v16h a = __builtin_shufflevector(pal[cur], pah[cur],
                                       0, 1, 2, 3, 4, 5, 6, 7,
                                       8, 9, 10, 11, 12, 13, 14, 15);
      v8f  c = __builtin_shufflevector(pcl[cur], pch[cur],
                                       0, 1, 2, 3, 4, 5, 6, 7);

      // Prefetch next block's fragments (wraps to block 0 for the next tile);
      // these fly during the wmma + tanh chain + LDS round trip below.
      const int nfb = (fb == 17) ? 0 : fb + 1;
      frag_issue(fragLane + (size_t)nfb * 2048,
                 pal[nxt], pah[nxt], pcl[nxt], pch[nxt]);

      c = __builtin_amdgcn_wmma_f32_16x16x32_f16(
              false, a, false, bvec, (short)0, c, false, false);

      if (L < 14) {
        // tanh + cvt to f16, store 8 contiguous features back to LDS
        v8h o;
#pragma unroll
        for (int r = 0; r < 8; ++r)
          o[r] = (_Float16)my_tanh(c[r]);
        *(v8h*)(hrow + 16 * mb + 8 * lh) = o;
      } else {
        // final layer 16->3, no tanh; rows M=0..2 live in lanes 0-15, vgpr 0..2
        if (lh == 0) {
          out[0 * NPTS + p] = c[0];
          out[1 * NPTS + p] = c[1];
          out[2 * NPTS + p] = c[2];
        }
      }
    }
  }
  // Dangling prefetch from the last tile is absorbed by S_ENDPGM's wait-idle.
}

extern "C" void kernel_launch(void* const* d_in, const int* in_sizes, int n_in,
                              void* d_out, int out_size, void* d_ws, size_t ws_size,
                              hipStream_t stream) {
  // d_in order (dict insertion): coords, W0, b0, W1, b1, ..., W14, b14
  const float* coords = (const float*)d_in[0];
  const float* W[15];
  const float* B[15];
  for (int i = 0; i < 15; ++i) {
    W[i] = (const float*)d_in[1 + 2 * i];
    B[i] = (const float*)d_in[2 + 2 * i];
  }
  char* wsF = (char*)d_ws;  // 18 * 2048 = 36864 B

  pinn_prep<<<18, 32, 0, stream>>>(
      W[0], W[1], W[2], W[3], W[4], W[5], W[6], W[7], W[8], W[9],
      W[10], W[11], W[12], W[13], W[14],
      B[0], B[1], B[2], B[3], B[4], B[5], B[6], B[7], B[8], B[9],
      B[10], B[11], B[12], B[13], B[14],
      wsF);

  pinn_main<<<1024, 256, 0, stream>>>(coords, wsF, (float*)d_out);
}
